// HeadMaskAttn_2525440770360
// MI455X (gfx1250) — matrix operand order repaired and checked by reading, not compiled, not run on hardware
//
#include <hip/hip_runtime.h>

#define HH 12
#define DD 64
#define NN 1024
#define BB 8
#define C3 2304      // 3*H*D channels in x
#define OC 768       // H*D output channels
#define MAXREL 14
#define NTAB 29      // 2*MAXREL+1
#define SCALEF 0.125f // 64^-0.5

typedef __attribute__((ext_vector_type(16))) _Float16 v16h;
typedef __attribute__((ext_vector_type(8)))  _Float16 v8h;
typedef __attribute__((ext_vector_type(4)))  _Float16 v4h;
typedef __attribute__((ext_vector_type(8)))  float    v8f;

__device__ __forceinline__ v8f wmma32(v16h a, v16h b, v8f c) {
  // D = A(16x32 f16) * B(32x16 f16) + C(16x16 f32)
  return __builtin_amdgcn_wmma_f32_16x16x32_f16(false, a, false, b, (short)0, c,
                                                false, false);
}

// Build a 16-half fragment from two 8-half (16B) chunks in LDS.
__device__ __forceinline__ v16h ld16(const _Float16* lo_p, const _Float16* hi_p) {
  v8h lo = *(const v8h*)lo_p;
  v8h hi = *(const v8h*)hi_p;
  v16h r;
#pragma unroll
  for (int i = 0; i < 8; ++i) { r[i] = lo[i]; r[i + 8] = hi[i]; }
  return r;
}

__device__ __forceinline__ int clipdist(int d) {
  d = (d < -MAXREL) ? -MAXREL : d;
  d = (d >  MAXREL) ?  MAXREL : d;
  return d + MAXREL;
}

__global__ __launch_bounds__(128) void attn_flash_wmma(
    const float* __restrict__ x,
    const float* __restrict__ rkt,
    const float* __restrict__ rvt,
    float* __restrict__ out)
{
  // LDS (51456 B total)
  __shared__ _Float16 k_tile[64][72];      // [key][d]   row-major, padded
  __shared__ _Float16 vT_tile[64][72];     // [d][key]   transposed, padded
  __shared__ _Float16 p_stage[4][16][72];  // per-wave P round-trip
  __shared__ float    qrel_s[4][16][32];   // per-wave SCALE * q . rel_k_table[t]
  __shared__ float    asum_s[4][16][32];   // per-wave bucketed attn sums
  __shared__ float    rvt_s[NTAB][64];     // rel_v_table staged

  const int tid  = threadIdx.x;
  const int wv   = tid >> 5;
  const int lane = tid & 31;
  const int lrow = lane & 15;   // M / N index within 16
  const int hs   = lane >> 4;   // half-wave select

  const int qt = blockIdx.x & 15;          // 16 query tiles of 64 rows
  const int h  = (blockIdx.x >> 4) % HH;
  const int b  = blockIdx.x / (16 * HH);
  const int q0 = qt * 64 + wv * 16;        // this wave's first query row

  // Stage rel_v table (read in epilogue, covered by loop barriers)
  for (int i = tid; i < NTAB * 64; i += 128) rvt_s[i >> 6][i & 63] = rvt[i];

  // Zero per-wave bucket accumulators (wave-local, LDS in-order)
#pragma unroll
  for (int r = 0; r < 16; ++r) asum_s[wv][r][lane] = 0.f;

  // This lane's query row (q channel block: s=0 -> offset h*64)
  const float* qp = x + (size_t)(b * NN + q0 + lrow) * C3 + h * DD;

  // qrel[row][t] = SCALE * q[row] . rel_k_table[t]   (halves split t odd/even)
  for (int t = hs; t < NTAB; t += 2) {
    const float* tb = rkt + t * DD;
    float acc = 0.f;
#pragma unroll
    for (int d = 0; d < DD; ++d) acc += qp[d] * tb[d];
    qrel_s[wv][lrow][t] = acc * SCALEF;
  }

  // Q A-fragments with SCALE folded in.
  // A 16x32 f16 layout: lane holds row=lrow; halves[0..7]=K hs*8+0..7,
  // halves[8..15]=K 16+hs*8+0..7; fragment f covers d = f*32 .. f*32+31.
  v16h qa[2];
#pragma unroll
  for (int f = 0; f < 2; ++f) {
    const float* qb = qp + f * 32 + hs * 8;
#pragma unroll
    for (int j = 0; j < 8; ++j) {
      qa[f][j]     = (_Float16)(qb[j]      * SCALEF);
      qa[f][j + 8] = (_Float16)(qb[16 + j] * SCALEF);
    }
  }

  // Online-softmax state: rows r+8*hs, cols = lrow in C/D layout
  v8f o[4];
#pragma unroll
  for (int dt = 0; dt < 4; ++dt) o[dt] = {};
  float mrow[8], lsum[8];
#pragma unroll
  for (int r = 0; r < 8; ++r) { mrow[r] = -1e30f; lsum[r] = 0.f; }

  const float* kg = x + (size_t)(b * NN) * C3 + (HH + h) * DD;      // k: s=1
  const float* vg = x + (size_t)(b * NN) * C3 + (2 * HH + h) * DD;  // v: s=2

  for (int kt = 0; kt < 16; ++kt) {
    const int kbase = kt * 64;

    // Cooperative stage: K row-major f32->f16, V transposed f32->f16.
#pragma unroll
    for (int i = 0; i < 8; ++i) {
      int idx = tid + i * 128;            // 0..1023
      int kr = idx >> 4, c4 = idx & 15;
      const float4 k4 = *(const float4*)(kg + (size_t)(kbase + kr) * C3 + c4 * 4);
      v4h kh = { (_Float16)k4.x, (_Float16)k4.y, (_Float16)k4.z, (_Float16)k4.w };
      *(v4h*)&k_tile[kr][c4 * 4] = kh;
      const float4 v4 = *(const float4*)(vg + (size_t)(kbase + kr) * C3 + c4 * 4);
      vT_tile[c4 * 4 + 0][kr] = (_Float16)v4.x;
      vT_tile[c4 * 4 + 1][kr] = (_Float16)v4.y;
      vT_tile[c4 * 4 + 2][kr] = (_Float16)v4.z;
      vT_tile[c4 * 4 + 3][kr] = (_Float16)v4.w;
      if (kt < 15) {  // prefetch next tile -> global_prefetch_b8
        __builtin_prefetch(kg + (size_t)(kbase + 64 + kr) * C3 + c4 * 4, 0, 1);
        __builtin_prefetch(vg + (size_t)(kbase + 64 + kr) * C3 + c4 * 4, 0, 1);
      }
    }
    __syncthreads();

    // S = (SCALE*Q) K^T : 4 col-tiles x 2 k-chunks of WMMA.
    // B 32x16 layout: lane holds col=lrow; halves[0..7] = K hs*8+0..7,
    //   halves[8..15] = K 16+hs*8+0..7 of K's row (key = nt*16+lrow).
    v8f s[4];
#pragma unroll
    for (int nt = 0; nt < 4; ++nt) {
      const _Float16* kp = &k_tile[nt * 16 + lrow][hs * 8];
      v8f acc = {};
      acc = wmma32(qa[0], ld16(kp,      kp + 16), acc);
      acc = wmma32(qa[1], ld16(kp + 32, kp + 48), acc);
      s[nt] = acc;
    }

    // Add relative-position bias, compute tile row-max.
    float mt[8];
#pragma unroll
    for (int r = 0; r < 8; ++r) mt[r] = -1e30f;
#pragma unroll
    for (int nt = 0; nt < 4; ++nt) {
      const int m_g = kbase + nt * 16 + lrow;
#pragma unroll
      for (int r = 0; r < 8; ++r) {
        const int row = r + 8 * hs;
        const int dd = clipdist(m_g - (q0 + row));
        float v = s[nt][r] + qrel_s[wv][row][dd];
        s[nt][r] = v;
        mt[r] = fmaxf(mt[r], v);
      }
    }
#pragma unroll
    for (int r = 0; r < 8; ++r)
#pragma unroll
      for (int msk = 1; msk <= 8; msk <<= 1)
        mt[r] = fmaxf(mt[r], __shfl_xor(mt[r], msk, 32));

    float alpha[8];
#pragma unroll
    for (int r = 0; r < 8; ++r) {
      float mn = fmaxf(mrow[r], mt[r]);
      alpha[r] = __expf(mrow[r] - mn);
      mrow[r] = mn;
    }
    // Rescale O accumulators and bucket sums by alpha.
#pragma unroll
    for (int dt = 0; dt < 4; ++dt)
#pragma unroll
      for (int r = 0; r < 8; ++r) o[dt][r] *= alpha[r];
#pragma unroll
    for (int r = 0; r < 8; ++r) {
      const int row = r + 8 * hs;
      asum_s[wv][row][lrow]      *= alpha[r];   // each (row,t) owned by 1 lane
      asum_s[wv][row][lrow + 16] *= alpha[r];
    }

    // P = exp(s - m): rowsum, stage f16 for PV, bucket-accumulate for rel-V.
    float rs[8];
#pragma unroll
    for (int r = 0; r < 8; ++r) rs[r] = 0.f;
#pragma unroll
    for (int nt = 0; nt < 4; ++nt) {
      const int m_g = kbase + nt * 16 + lrow;
#pragma unroll
      for (int r = 0; r < 8; ++r) {
        const int row = r + 8 * hs;
        float p = __expf(s[nt][r] - mrow[r]);
        rs[r] += p;
        p_stage[wv][row][nt * 16 + lrow] = (_Float16)p;
        atomicAdd(&asum_s[wv][row][clipdist(m_g - (q0 + row))], p);  // ds_add_f32
      }
    }
#pragma unroll
    for (int r = 0; r < 8; ++r) {
#pragma unroll
      for (int msk = 1; msk <= 8; msk <<= 1)
        rs[r] += __shfl_xor(rs[r], msk, 32);
      lsum[r] = lsum[r] * alpha[r] + rs[r];
    }

    // O += P V  (P re-read in A-frag layout; V B-frag contiguous via vT)
#pragma unroll
    for (int kc = 0; kc < 2; ++kc) {
      const _Float16* pp = &p_stage[wv][lrow][kc * 32 + hs * 8];
      v16h pa = ld16(pp, pp + 16);
#pragma unroll
      for (int dt = 0; dt < 4; ++dt) {
        const _Float16* vp = &vT_tile[dt * 16 + lrow][kc * 32 + hs * 8];
        o[dt] = wmma32(pa, ld16(vp, vp + 16), o[dt]);
      }
    }
    __syncthreads();
  }

  // Epilogue: out = (O + asum @ rel_v_table) / l
#pragma unroll
  for (int r = 0; r < 8; ++r) {
    const int row = r + 8 * hs;
    float relv[4] = {0.f, 0.f, 0.f, 0.f};
    for (int t = 0; t < NTAB; ++t) {
      float a = asum_s[wv][row][t];
#pragma unroll
      for (int dt = 0; dt < 4; ++dt)
        relv[dt] += a * rvt_s[t][dt * 16 + lrow];
    }
    const float inv = 1.0f / lsum[r];
    float* op = out + (size_t)(b * NN + q0 + row) * OC + h * DD + lrow;
#pragma unroll
    for (int dt = 0; dt < 4; ++dt)
      op[dt * 16] = (o[dt][r] + relv[dt]) * inv;
  }
}

extern "C" void kernel_launch(void* const* d_in, const int* in_sizes, int n_in,
                              void* d_out, int out_size, void* d_ws, size_t ws_size,
                              hipStream_t stream) {
  const float* x   = (const float*)d_in[0];
  const float* rkt = (const float*)d_in[1];
  const float* rvt = (const float*)d_in[2];
  float* out = (float*)d_out;
  (void)in_sizes; (void)n_in; (void)out_size; (void)d_ws; (void)ws_size;

  dim3 grid(BB * HH * 16);   // 1536 workgroups: (b, h, 64-row query tile)
  dim3 block(128);           // 4 wave32 waves, one 16-row tile each
  hipLaunchKernelGGL(attn_flash_wmma, grid, block, 0, stream, x, rkt, rvt, out);
}
